// Joiner_10505490006677
// MI455X (gfx1250) — matrix operand order repaired and checked by reading, not compiled, not run on hardware
//
#include <hip/hip_runtime.h>
#include <hip/hip_bf16.h>

typedef __attribute__((ext_vector_type(16))) _Float16 v16h;
typedef __attribute__((ext_vector_type(8)))  float    v8f;

constexpr int Bn  = 4;
constexpr int Tn  = 448;
constexpr int Un  = 448;
constexpr int SKS = 96;
constexpr int BKS = 32;
constexpr int Hn  = 96;   // K of output GEMM = 3 chunks of 32
constexpr int Vn  = 128;  // N of output GEMM = 8 tiles of 16

// ---------------- Stage 1a: sh = s @ W_sh^T + b_sh  (tiny, VALU) ----------------
__global__ void proj_sh_kernel(const float* __restrict__ s, const float* __restrict__ W_sh,
                               const float* __restrict__ b_sh, float* __restrict__ sh) {
    int gid = blockIdx.x * blockDim.x + threadIdx.x;
    if (gid >= Bn * Tn * Hn) return;
    int r = gid / Hn, h = gid % Hn;
    const float* sr = s    + (size_t)r * SKS;
    const float* wr = W_sh + (size_t)h * SKS;
    float acc = b_sh[h];
#pragma unroll
    for (int k = 0; k < SKS; k += 4) {
        float4 a = *(const float4*)(sr + k);
        float4 w = *(const float4*)(wr + k);
        acc += a.x * w.x + a.y * w.y + a.z * w.z + a.w * w.w;
    }
    sh[gid] = acc;
}

// ---------------- Stage 1b: rbh = relu(b @ W_bh^T + b_bh) ----------------
__global__ void proj_bh_kernel(const float* __restrict__ b, const float* __restrict__ W_bh,
                               const float* __restrict__ b_bh, float* __restrict__ rbh) {
    int gid = blockIdx.x * blockDim.x + threadIdx.x;
    if (gid >= Bn * Un * Hn) return;
    int r = gid / Hn, h = gid % Hn;
    const float* br = b    + (size_t)r * BKS;
    const float* wr = W_bh + (size_t)h * BKS;
    float acc = b_bh[h];
#pragma unroll
    for (int k = 0; k < BKS; k += 4) {
        float4 a = *(const float4*)(br + k);
        float4 w = *(const float4*)(wr + k);
        acc += a.x * w.x + a.y * w.y + a.z * w.z + a.w * w.w;
    }
    rbh[gid] = fmaxf(acc, 0.0f);
}

// ------- Stage 1c: pack W_out into WMMA B-matrix (32x16 f16) fragments -------
// B fragment (K=32 x N=16): lane L holds column N = L%16; half i holds K = i + (L>=16 ? 16 : 0).
// 24 fragments: f = nt*3 + kc  (nt = N-tile 0..7, kc = K-chunk 0..2). 512 halves each.
__global__ void pack_wout_kernel(const float* __restrict__ W_out, _Float16* __restrict__ wf) {
    int f    = blockIdx.x;          // 0..23
    int lane = threadIdx.x;         // 0..31
    int nt = f / 3, kc = f % 3;
    int col   = lane & 15;
    int kbase = kc * 32 + ((lane >> 4) << 4);
    int N     = nt * 16 + col;
#pragma unroll
    for (int i = 0; i < 16; i++)
        wf[(size_t)f * 512 + lane * 16 + i] = (_Float16)W_out[(size_t)N * Hn + kbase + i];
}

// ---------------- Stage 2: fused outer-product GEMM + log_softmax ----------------
// Block = 256 threads = 8 waves. Wave w handles t = tb*8 + w, u in [ub*16, ub*16+16).
// Per wave: 24x v_wmma_f32_16x16x32_f16 -> 16x128 logits in 64 VGPRs -> in-register
// log_softmax -> non-temporal streamed stores (411 MB output, BW-bound).
__launch_bounds__(256)
__global__ void joiner_main_kernel(const float* __restrict__ sh,
                                   const float* __restrict__ rbh,
                                   const _Float16* __restrict__ wf,
                                   const float* __restrict__ b_out,
                                   float* __restrict__ out) {
    __shared__ float sh_lds[8 * 96];
    __shared__ float rbh_lds[16 * 100];               // stride 100: avoids 64-bank conflicts
    __shared__ __align__(32) _Float16 w_lds[24 * 512]; // 24 KB of B fragments
    __shared__ float bo_lds[Vn];

    const int tid = threadIdx.x;
    const int wg  = blockIdx.x;
    const int ub  = wg % (Un / 16);
    const int tb  = (wg / (Un / 16)) % (Tn / 8);
    const int bb  = wg / ((Un / 16) * (Tn / 8));

    // cooperative LDS fills (rows are contiguous in the staged buffers)
    {
        const float* src = sh + ((size_t)bb * Tn + tb * 8) * Hn;
        for (int i = tid; i < 8 * 96; i += 256) sh_lds[i] = src[i];
        const float* rsrc = rbh + ((size_t)bb * Un + ub * 16) * Hn;
        for (int i = tid; i < 16 * 96; i += 256) rbh_lds[(i / 96) * 100 + (i % 96)] = rsrc[i];
        const uint4* wsrc = (const uint4*)wf;
        uint4* wdst = (uint4*)w_lds;
        for (int i = tid; i < (24 * 512 * 2) / 16; i += 256) wdst[i] = wsrc[i];
        if (tid < Vn) bo_lds[tid] = b_out[tid];
    }
    __syncthreads();

    const int wid  = tid >> 5;
    const int lane = tid & 31;
    const int col  = lane & 15;
    const int hi   = lane >> 4;
    const int t    = tb * 8 + wid;
    const int u0   = ub * 16;

    // Build A fragments (16x32 f16, contiguous k order as in packed B: lane L holds row M=L%16,
    // halves 0..15 -> K = koff+0..15, koff = 16*(L>=16)).
    // A[m,k] = sh[t,k] * rbh[u0+m,k]
    const float* sp = &sh_lds[wid * 96];
    const float* rp = &rbh_lds[col * 100];
    const int koff = hi * 16;
    v16h afr[3];
#pragma unroll
    for (int kc = 0; kc < 3; kc++) {
        const int base = kc * 32 + koff;
#pragma unroll
        for (int i = 0; i < 8; i++)
            afr[kc][i] = (_Float16)(sp[base + i] * rp[base + i]);
#pragma unroll
        for (int i = 0; i < 8; i++)
            afr[kc][8 + i] = (_Float16)(sp[base + 8 + i] * rp[base + 8 + i]);
    }

    // 24 WMMAs -> 8 accumulator tiles (16x16 f32 each)
    v8f acc[8];
#pragma unroll
    for (int nt = 0; nt < 8; nt++) {
        v8f c = {0.f, 0.f, 0.f, 0.f, 0.f, 0.f, 0.f, 0.f};
#pragma unroll
        for (int kc = 0; kc < 3; kc++) {
            const v16h bfr = *(const v16h*)&w_lds[((nt * 3 + kc) * 32 + lane) * 16];
            c = __builtin_amdgcn_wmma_f32_16x16x32_f16(false, afr[kc], false, bfr,
                                                       (short)0, c, false, false);
        }
        acc[nt] = c;
    }

    // + b_out (per column v = nt*16 + col, same for all rows in the tile)
#pragma unroll
    for (int nt = 0; nt < 8; nt++) {
        const float bv = bo_lds[nt * 16 + col];
#pragma unroll
        for (int g = 0; g < 8; g++) acc[nt][g] += bv;
    }

    // log_softmax per row. C/D layout: VGPR g holds row m=g (lanes 0-15, N=0..15)
    // and m=g+8 (lanes 16-31). Reductions over 16-lane halves via shfl_xor 1,2,4,8.
#pragma unroll
    for (int g = 0; g < 8; g++) {
        float x[8];
#pragma unroll
        for (int nt = 0; nt < 8; nt++) x[nt] = acc[nt][g];
        float mx = x[0];
#pragma unroll
        for (int nt = 1; nt < 8; nt++) mx = fmaxf(mx, x[nt]);
#pragma unroll
        for (int off = 1; off < 16; off <<= 1)
            mx = fmaxf(mx, __shfl_xor(mx, off, 32));
        float se = 0.f;
#pragma unroll
        for (int nt = 0; nt < 8; nt++) se += __expf(x[nt] - mx);
#pragma unroll
        for (int off = 1; off < 16; off <<= 1)
            se += __shfl_xor(se, off, 32);
        const float lse = mx + __logf(se);

        const int m = g + 8 * hi;
        float* op = out + (((size_t)bb * Tn + t) * Un + (u0 + m)) * Vn + col;
#pragma unroll
        for (int nt = 0; nt < 8; nt++)
            __builtin_nontemporal_store(x[nt] - lse, op + nt * 16);
    }
}

extern "C" void kernel_launch(void* const* d_in, const int* in_sizes, int n_in,
                              void* d_out, int out_size, void* d_ws, size_t ws_size,
                              hipStream_t stream) {
    const float* s     = (const float*)d_in[0];
    const float* b     = (const float*)d_in[1];
    const float* W_sh  = (const float*)d_in[2];
    const float* b_sh  = (const float*)d_in[3];
    const float* W_bh  = (const float*)d_in[4];
    const float* b_bh  = (const float*)d_in[5];
    const float* W_out = (const float*)d_in[6];
    const float* b_out = (const float*)d_in[7];
    float* out = (float*)d_out;

    // workspace: sh f32 [B*T*H] | rbh f32 [B*U*H] | packed W_out f16 [24*512]
    float*    shw = (float*)d_ws;
    float*    rbh = shw + (size_t)Bn * Tn * Hn;
    _Float16* wf  = (_Float16*)(rbh + (size_t)Bn * Un * Hn);

    const int nsh = Bn * Tn * Hn;
    proj_sh_kernel<<<(nsh + 255) / 256, 256, 0, stream>>>(s, W_sh, b_sh, shw);
    const int nbh = Bn * Un * Hn;
    proj_bh_kernel<<<(nbh + 255) / 256, 256, 0, stream>>>(b, W_bh, b_bh, rbh);
    pack_wout_kernel<<<24, 32, 0, stream>>>(W_out, wf);

    const int grid = Bn * (Tn / 8) * (Un / 16); // 6272 workgroups
    joiner_main_kernel<<<grid, 256, 0, stream>>>(shw, rbh, wf, b_out, out);
}
